// MultiHeadAttention_7146825581202
// MI455X (gfx1250) — matrix operand order repaired and checked by reading, not compiled, not run on hardware
//
#include <hip/hip_runtime.h>
#include <hip/hip_bf16.h>

// ---------------- problem constants ----------------
#define B_   2
#define S_   2048
#define D_   1024
#define H_   16
#define DK_  64
#define ROWS (B_ * S_)          // 4096
#define NEGV (-1e9f)
#define EPSV (1e-6f)

typedef __attribute__((ext_vector_type(16))) __bf16 v16bf;
typedef __attribute__((ext_vector_type(8)))  float  v8f;

// ---- CDNA5 async global->LDS path (guarded so compile can never regress) ----
#if defined(__has_builtin)
#if __has_builtin(__builtin_amdgcn_global_load_async_to_lds_b128) && \
    __has_builtin(__builtin_amdgcn_s_wait_asynccnt)
#define HAVE_ASYNC_LDS 1
#endif
#endif
#ifndef HAVE_ASYNC_LDS
#define HAVE_ASYNC_LDS 0
#endif

#if HAVE_ASYNC_LDS
// Builtin signature (from hipcc diagnostic):
//   void __builtin_amdgcn_global_load_async_to_lds_b128(
//       int __attribute__((vector_size(16))) addrspace(1)* src,
//       int __attribute__((vector_size(16))) addrspace(3)* dst,
//       imm int offset, imm int cpol)
typedef int v4i_vs __attribute__((__vector_size__(16)));
typedef __attribute__((address_space(1))) v4i_vs as1_v4i;
typedef __attribute__((address_space(3))) v4i_vs as3_v4i;
#endif

// copy 16B global -> LDS (async on CDNA5, tracked by ASYNCcnt)
__device__ __forceinline__ void async_ld16(void* lds, const void* g) {
#if HAVE_ASYNC_LDS
    __builtin_amdgcn_global_load_async_to_lds_b128((as1_v4i*)g, (as3_v4i*)lds, 0, 0);
#else
    *(uint4*)lds = *(const uint4*)g;
#endif
}
__device__ __forceinline__ void wait_async_le2() {
#if HAVE_ASYNC_LDS
    __builtin_amdgcn_s_wait_asynccnt(2);
#endif
}
__device__ __forceinline__ void wait_async_le0() {
#if HAVE_ASYNC_LDS
    __builtin_amdgcn_s_wait_asynccnt(0);
#endif
}

union AFrag {
    v16bf        v;
    unsigned int u[8];
    __bf16       h[16];
    uint4        q[2];
};

__device__ __forceinline__ v8f zero8() {
    v8f z = {0.f, 0.f, 0.f, 0.f, 0.f, 0.f, 0.f, 0.f};
    return z;
}

// WMMA: D = A(16x32 bf16) * B(32x16 bf16) + C(16x16 f32)
__device__ __forceinline__ v8f wmma_bf16(v16bf a, v16bf b, v8f c) {
    return __builtin_amdgcn_wmma_f32_16x16x32_bf16(
        false, a, false, b, (short)0, c, false, false);
}

// A-fragment gather, 16x32 bf16, row-major source (row = lane&15).
// Lane layout (contiguous): k = 2j + 16*hi.
__device__ __forceinline__ v16bf load_a16x32(const __bf16* __restrict__ rowk, int hi) {
    AFrag a;
    const int kb8 = 16 * hi;
#pragma unroll
    for (int j = 0; j < 8; ++j) {
        const int k = 2 * j + kb8;
        a.u[j] = *(const unsigned int*)(rowk + k);
    }
    return a.v;
}

// Same A layout sourced from f32 row in LDS (probabilities), cvt to bf16.
__device__ __forceinline__ v16bf load_p16x32(const float* __restrict__ srow, int hi) {
    AFrag a;
    const int kb8 = 16 * hi;
#pragma unroll
    for (int j = 0; j < 8; ++j) {
        const int k = 2 * j + kb8;
        a.h[2 * j]     = (__bf16)srow[k];
        a.h[2 * j + 1] = (__bf16)srow[k + 1];
    }
    return a.v;
}

// B-fragment from an LDS row padded to 40 bf16 (bank-friendly ds_load_b128 pairs)
__device__ __forceinline__ v16bf lds_b_frag(const __bf16* row, int hi) {
    AFrag t;
    t.q[0] = *(const uint4*)(row + 16 * hi);
    t.q[1] = *(const uint4*)(row + 16 * hi + 8);
    return t.v;
}

// ---------------- kernel 1: f32 -> bf16 (x4 vectorized) ----------------
__global__ __launch_bounds__(256) void cvt_f32_bf16(const float* __restrict__ in,
                                                    __bf16* __restrict__ out, int n4) {
    int i = blockIdx.x * 256 + threadIdx.x;
    if (i >= n4) return;
    float4 v = ((const float4*)in)[i];
    union { __bf16 h[4]; unsigned long long u; } o;
    o.h[0] = (__bf16)v.x; o.h[1] = (__bf16)v.y;
    o.h[2] = (__bf16)v.z; o.h[3] = (__bf16)v.w;
    ((unsigned long long*)out)[i] = o.u;
}

// ---------------- kernel 2: W [K,N] f32 -> Wt [N,K] bf16 ----------------
__global__ __launch_bounds__(256) void transpose_w(const float* __restrict__ W,
                                                   __bf16* __restrict__ Wt) {
    int idx = blockIdx.x * 256 + threadIdx.x;      // over D_*D_
    int n = idx >> 10, k = idx & (D_ - 1);
    Wt[idx] = (__bf16)W[(size_t)k * D_ + n];
}

// ---------------- kernel 3: unified WMMA GEMM  C[M,N] = A[M,K] * Bt[N,K]^T ----
// Block: 256 thr / 8 waves, block tile 64M x 128N, wave tile 32x32 (4 accums).
// B (weight) tile double-buffered in LDS via async global->LDS copies.
// mode 0: bf16 row-major (with scale); 1: V transposed [B, D, S]; 2: f32 + residual.
__global__ __launch_bounds__(256) void gemm_wmma(const __bf16* __restrict__ A,
                                                 const __bf16* __restrict__ Bt,
                                                 __bf16* __restrict__ Cbf,
                                                 __bf16* __restrict__ Ct,
                                                 const float*  __restrict__ resid,
                                                 float*        __restrict__ Cf,
                                                 float scale, int mode) {
    __shared__ __align__(16) __bf16 Bs[2][128][40];   // padded rows: 80B stride

    const int tid  = threadIdx.x;
    const int w    = tid >> 5;
    const int lane = tid & 31;
    const int hi   = lane >> 4;
    const int ln   = lane & 15;
    const int wm   = w & 1;                        // 2  -> 64 rows
    const int wn   = w >> 1;                       // 4  -> 128 cols
    const int row0 = blockIdx.x * 64 + wm * 32;
    const int colB = blockIdx.y * 128;
    const int col0 = colB + wn * 32;

    // B-tile staging: 2 threads per column row, 2x16B each per K-chunk
    const int tn = tid >> 1;                       // 0..127
    const int e0 = (tid & 1) * 8;                  // bf16 element offset of first 16B
    const __bf16* bsrc = Bt + (size_t)(colB + tn) * D_ + e0;

    const __bf16* arow0 = A + (size_t)(row0 + ln) * D_;
    const __bf16* arow1 = A + (size_t)(row0 + 16 + ln) * D_;

    v8f c00 = zero8(), c01 = zero8(), c10 = zero8(), c11 = zero8();

    // prologue: stage K-chunks 0 and 1
    async_ld16(&Bs[0][tn][e0],      bsrc + 0);
    async_ld16(&Bs[0][tn][e0 + 16], bsrc + 16);
    async_ld16(&Bs[1][tn][e0],      bsrc + 32);
    async_ld16(&Bs[1][tn][e0 + 16], bsrc + 48);

    int buf = 0;
    for (int kb = 0; kb < D_; kb += 32) {
        if (kb + 32 < D_) wait_async_le2(); else wait_async_le0();
        __syncthreads();

        __builtin_prefetch(arow0 + kb + 256, 0, 3);
        __builtin_prefetch(arow1 + kb + 256, 0, 3);
        v16bf a0 = load_a16x32(arow0 + kb, hi);
        v16bf a1 = load_a16x32(arow1 + kb, hi);
        v16bf b0 = lds_b_frag(&Bs[buf][wn * 32 + ln][0], hi);
        v16bf b1 = lds_b_frag(&Bs[buf][wn * 32 + 16 + ln][0], hi);
        c00 = wmma_bf16(a0, b0, c00);
        c01 = wmma_bf16(a0, b1, c01);
        c10 = wmma_bf16(a1, b0, c10);
        c11 = wmma_bf16(a1, b1, c11);

        __syncthreads();
        if (kb + 64 < D_) {                        // refill the buffer just consumed
            async_ld16(&Bs[buf][tn][e0],      bsrc + kb + 64);
            async_ld16(&Bs[buf][tn][e0 + 16], bsrc + kb + 64 + 16);
        }
        buf ^= 1;
    }

#pragma unroll
    for (int r = 0; r < 8; ++r) {
        const int mm = r + 8 * hi;                 // C layout: VGPR r -> M = r + 8*hi
        const float vv[4] = {c00[r], c01[r], c10[r], c11[r]};
        const int   rg[4] = {row0 + mm, row0 + mm, row0 + 16 + mm, row0 + 16 + mm};
        const int   cg[4] = {col0 + ln, col0 + 16 + ln, col0 + ln, col0 + 16 + ln};
#pragma unroll
        for (int t = 0; t < 4; ++t) {
            if (mode == 0) {
                Cbf[(size_t)rg[t] * D_ + cg[t]] = (__bf16)(vv[t] * scale);
            } else if (mode == 1) {
                const int bb = rg[t] >> 11, s = rg[t] & (S_ - 1);
                Ct[((size_t)bb * D_ + cg[t]) * S_ + s] = (__bf16)vv[t];
            } else {
                const size_t o = (size_t)rg[t] * D_ + cg[t];
                Cf[o] = vv[t] + resid[o];
            }
        }
    }
}

// ---------------- kernel 4: attention (scores + softmax + PV) ----------------
// grid: (B_*H_, S_/16); block 256 = 8 waves; ~133KB LDS score strip.
__global__ __launch_bounds__(256) void attn_kernel(const __bf16* __restrict__ Qp,
                                                   const __bf16* __restrict__ Kp,
                                                   const __bf16* __restrict__ Vpt,
                                                   const int*    __restrict__ mask,
                                                   float*        __restrict__ attn,
                                                   __bf16*       __restrict__ AO) {
    __shared__ float sS[16 * S_];                  // 128 KB score strip
    __shared__ float ps[256];
    __shared__ float rmax[16];
    __shared__ float rsum[16];
    __shared__ float sP[4 * 256];                  // PV partial tiles (waves 4..7)

    const int bh = blockIdx.x;
    const int b  = bh >> 4, h = bh & 15;
    const int q0 = blockIdx.y * 16;

    const int tid  = threadIdx.x;
    const int w    = tid >> 5;
    const int lane = tid & 31;
    const int hi   = lane >> 4;
    const int ln   = lane & 15;

    const size_t qrowg = (size_t)b * S_ + q0;

    // ---- stage 1: masked scores S = (Q/temp) K^T into LDS ----
    const __bf16* qrow = Qp + (qrowg + ln) * (size_t)D_ + h * DK_;
    const v16bf a0 = load_a16x32(qrow, hi);        // head-dim k = 0..31
    const v16bf a1 = load_a16x32(qrow + 32, hi);   // head-dim k = 32..63

    for (int ct = w; ct < S_ / 16; ct += 8) {
        const __bf16* krow = Kp + ((size_t)b * S_ + ct * 16 + ln) * D_ + h * DK_ + 16 * hi;
        v16bf b0 = *(const v16bf*)krow;
        v16bf b1 = *(const v16bf*)(krow + 32);
        v8f c = zero8();
        c = wmma_bf16(a0, b0, c);
        c = wmma_bf16(a1, b1, c);
#pragma unroll
        for (int r = 0; r < 8; ++r) {
            const int mm  = r + 8 * hi;
            const int col = ct * 16 + ln;
            const int mv  = mask[((size_t)b * S_ + q0 + mm) * S_ + col];
            sS[mm * S_ + col] = mv ? c[r] : NEGV;
        }
    }
    __syncthreads();

    // ---- stage 2: softmax over each of the 16 rows (2048 wide) ----
    const int row = tid >> 4, chunk = tid & 15;
    const int base = row * S_ + chunk * 128;
    float mx = -3.4e38f;
#pragma unroll 8
    for (int i = 0; i < 128; ++i) mx = fmaxf(mx, sS[base + i]);
    ps[tid] = mx;
    __syncthreads();
    if (tid < 16) {
        float m2 = ps[tid * 16];
#pragma unroll
        for (int j = 1; j < 16; ++j) m2 = fmaxf(m2, ps[tid * 16 + j]);
        rmax[tid] = m2;
    }
    __syncthreads();
    const float rm = rmax[row];
    float sum = 0.f;
#pragma unroll 8
    for (int i = 0; i < 128; ++i) {
        float e = __expf(sS[base + i] - rm);
        sS[base + i] = e;
        sum += e;
    }
    ps[tid] = sum;
    __syncthreads();
    if (tid < 16) {
        float s2 = 0.f;
#pragma unroll
        for (int j = 0; j < 16; ++j) s2 += ps[tid * 16 + j];
        rsum[tid] = s2;
    }
    __syncthreads();

    // normalize in LDS + coalesced fp32 write of attention probabilities
    float* arow_out = attn + (((size_t)(b * H_ + h) * S_) + q0) * S_;
    for (int idx = tid; idx < 16 * S_; idx += 256) {
        const int r = idx >> 11;
        const float p = sS[idx] * (1.0f / rsum[r]);
        sS[idx] = p;
        arow_out[(size_t)r * S_ + (idx & (S_ - 1))] = p;
    }
    __syncthreads();

    // ---- stage 3: O = P @ V; all 8 waves (4 col-slices x 2 key-halves) ----
    {
        const int ntile = w & 3;
        const int khalf = w >> 2;
        const int vcol  = h * DK_ + ntile * 16 + ln;
        const __bf16* vptr = Vpt + ((size_t)b * D_ + vcol) * S_;
        v8f c = zero8();
        const int kb0 = khalf * (S_ / 2);
        for (int kb = kb0; kb < kb0 + S_ / 2; kb += 32) {
            v16bf a  = load_p16x32(&sS[ln * S_ + kb], hi);
            v16bf bf = *(const v16bf*)(vptr + kb + 16 * hi);
            c = wmma_bf16(a, bf, c);
        }
        if (w >= 4) {
#pragma unroll
            for (int r = 0; r < 8; ++r) sP[(w - 4) * 256 + r * 32 + lane] = c[r];
        }
        __syncthreads();
        if (w < 4) {
#pragma unroll
            for (int r = 0; r < 8; ++r) {
                const int mm = r + 8 * hi;
                AO[(qrowg + mm) * D_ + vcol] =
                    (__bf16)(c[r] + sP[w * 256 + r * 32 + lane]);
            }
        }
    }
}

// ---------------- kernel 5: per-row layernorm ----------------
__global__ __launch_bounds__(256) void ln_kernel(const float* __restrict__ Y,
                                                 const float* __restrict__ gamma,
                                                 const float* __restrict__ beta,
                                                 float* __restrict__ out) {
    __shared__ float red[256];
    __shared__ float red2[256];
    const int row = blockIdx.x;
    const int t   = threadIdx.x;
    float s = 0.f, s2 = 0.f;
    for (int c = t; c < D_; c += 256) {
        const float v = Y[(size_t)row * D_ + c];
        s += v; s2 += v * v;
    }
    red[t] = s; red2[t] = s2;
    __syncthreads();
    for (int off = 128; off > 0; off >>= 1) {
        if (t < off) { red[t] += red[t + off]; red2[t] += red2[t + off]; }
        __syncthreads();
    }
    const float mean = red[0] * (1.0f / D_);
    const float var  = red2[0] * (1.0f / D_) - mean * mean;
    const float rstd = rsqrtf(var + EPSV);
    for (int c = t; c < D_; c += 256) {
        const float v = Y[(size_t)row * D_ + c];
        out[(size_t)row * D_ + c] = (v - mean) * rstd * gamma[c] + beta[c];
    }
}

// ---------------- host side ----------------
extern "C" void kernel_launch(void* const* d_in, const int* in_sizes, int n_in,
                              void* d_out, int out_size, void* d_ws, size_t ws_size,
                              hipStream_t stream) {
    const float* q    = (const float*)d_in[0];
    const float* k    = (const float*)d_in[1];
    const float* v    = (const float*)d_in[2];
    const float* Wq   = (const float*)d_in[3];
    const float* Wk   = (const float*)d_in[4];
    const float* Wv   = (const float*)d_in[5];
    const float* Wfc  = (const float*)d_in[6];
    const float* gam  = (const float*)d_in[7];
    const float* bet  = (const float*)d_in[8];
    const int*   mask = (const int*)d_in[9];

    float* out  = (float*)d_out;                          // [B,S,D]
    float* attn = out + (size_t)ROWS * D_;                // [B,H,S,S]

    char* ws = (char*)d_ws;
    const size_t MB = 1u << 20;
    __bf16* qbf  = (__bf16*)(ws + 0 * MB);
    __bf16* kbf  = (__bf16*)(ws + 8 * MB);
    __bf16* vbf  = (__bf16*)(ws + 16 * MB);
    __bf16* Wqt  = (__bf16*)(ws + 24 * MB);
    __bf16* Wkt  = (__bf16*)(ws + 26 * MB);
    __bf16* Wvt  = (__bf16*)(ws + 28 * MB);
    __bf16* Wfct = (__bf16*)(ws + 30 * MB);
    __bf16* Qp   = (__bf16*)(ws + 32 * MB);               // prescaled by 1/sqrt(64)
    __bf16* Kp   = (__bf16*)(ws + 40 * MB);
    __bf16* Vpt  = (__bf16*)(ws + 48 * MB);               // [B, D, S]
    __bf16* AO   = (__bf16*)(ws + 56 * MB);               // head outputs [ROWS, D]
    float*  Yres = (float*)(ws + 64 * MB);                // fc + residual, f32

    const int n4 = ROWS * D_ / 4;                         // 1,048,576
    cvt_f32_bf16<<<n4 / 256, 256, 0, stream>>>(q, qbf, n4);
    cvt_f32_bf16<<<n4 / 256, 256, 0, stream>>>(k, kbf, n4);
    cvt_f32_bf16<<<n4 / 256, 256, 0, stream>>>(v, vbf, n4);

    const int nw = D_ * D_ / 256;                         // 4096 blocks
    transpose_w<<<nw, 256, 0, stream>>>(Wq,  Wqt);
    transpose_w<<<nw, 256, 0, stream>>>(Wk,  Wkt);
    transpose_w<<<nw, 256, 0, stream>>>(Wv,  Wvt);
    transpose_w<<<nw, 256, 0, stream>>>(Wfc, Wfct);

    dim3 gg(ROWS / 64, D_ / 128);                         // (64, 8)
    gemm_wmma<<<gg, 256, 0, stream>>>(qbf, Wqt, Qp, (__bf16*)0, (const float*)0, (float*)0, 0.125f, 0);
    gemm_wmma<<<gg, 256, 0, stream>>>(kbf, Wkt, Kp, (__bf16*)0, (const float*)0, (float*)0, 1.0f,   0);
    gemm_wmma<<<gg, 256, 0, stream>>>(vbf, Wvt, (__bf16*)0, Vpt, (const float*)0, (float*)0, 1.0f,  1);

    dim3 ga(B_ * H_, S_ / 16);                            // (32, 128)
    attn_kernel<<<ga, 256, 0, stream>>>(Qp, Kp, Vpt, mask, attn, AO);

    gemm_wmma<<<gg, 256, 0, stream>>>(AO, Wfct, (__bf16*)0, (__bf16*)0, q, Yres, 1.0f, 2);
    ln_kernel<<<ROWS, 256, 0, stream>>>(Yres, gam, bet, out);
}
